// LSTM_86577950753064
// MI455X (gfx1250) — compile-verified, matrix operand order checked by reading, not run on hardware
//
#include <hip/hip_runtime.h>
#include <hip/hip_bf16.h>
#include <stdint.h>

// ---------------------------------------------------------------------------
// CDNA5 (gfx1250) LSTM: bf16 WMMA GEMMs (double-buffered LDS pipeline) +
// persistent grid-synced recurrence. No lambdas / pointer arrays in device
// code: all staging lives in named scalars so nothing spills to scratch.
// ---------------------------------------------------------------------------

typedef __attribute__((ext_vector_type(16))) __bf16 v16bf;
typedef __attribute__((ext_vector_type(8)))  float  v8f;

#define LPITCH 40  // LDS row pitch in bf16 elems (32 data + 8 pad = 80B) -> conflict-free

__device__ __forceinline__ unsigned short f2bf(float f) {
  unsigned u = __builtin_bit_cast(unsigned, f);
  unsigned r = 0x7FFFu + ((u >> 16) & 1u);          // round-to-nearest-even
  return (unsigned short)((u + r) >> 16);
}
__device__ __forceinline__ float bf2f(unsigned short h) {
  unsigned u = ((unsigned)h) << 16;
  return __builtin_bit_cast(float, u);
}

__device__ __forceinline__ uint4 ldg4(const unsigned short* p) {
  return *(const uint4*)p;
}
__device__ __forceinline__ void st4(unsigned short* p, uint4 v) {
  *(uint4*)p = v;
}

// A/B 16-bit fragment gather per ISA 7.12.2: lane holds K=half*8..+7 and 16+half*8..+7
// from a [16 x 32] bf16 tile row (row ptr must be 16B aligned).
__device__ __forceinline__ v16bf load_frag(const unsigned short* row, int lhalf) {
  union { uint4 u[2]; v16bf v; } f;
  const uint4* p = (const uint4*)row;
  f.u[0] = p[lhalf];
  f.u[1] = p[2 + lhalf];
  return f.v;
}

__device__ __forceinline__ v8f wmma_bf16(v16bf a, v16bf b, v8f c) {
  return __builtin_amdgcn_wmma_f32_16x16x32_bf16(false, a, false, b, (short)0, c,
                                                 false, false);
}

// ---------------------------------------------------------------------------
// f32 -> bf16 conversion
// ---------------------------------------------------------------------------
__global__ void cvt_bf16_kernel(const float* __restrict__ in,
                                unsigned short* __restrict__ out, long long n) {
  long long i = (long long)blockIdx.x * blockDim.x + threadIdx.x;
  long long stride = (long long)gridDim.x * blockDim.x;
  for (; i < n; i += stride) out[i] = f2bf(in[i]);
}

// ---------------------------------------------------------------------------
// C[RxN] = A[RxK](bf16) * W[NxK]^T(bf16) (+bias).
// 256 thr = 8 waves; block tile 128x128; wave tile 32x64 (8 WMMA accums).
// Double-buffered LDS, register-staged global prefetch, one barrier/chunk,
// last K-chunk peeled so the steady-state prefetch is unconditional.
// ---------------------------------------------------------------------------
#define GT_BM 128
#define GT_BN 128

#define GEMM_LOAD(k0)                                   \
  ra0 = okA0 ? ldg4(Ab0 + (k0)) : uz;                   \
  ra1 = okA1 ? ldg4(Ab1 + (k0)) : uz;                   \
  rb0 = ldg4(Wb0 + (k0));                               \
  rb1 = ldg4(Wb1 + (k0));

#define GEMM_STORE(buf)                                 \
  st4(lA[buf] + sr0 * LPITCH + scg, ra0);               \
  st4(lA[buf] + sr1 * LPITCH + scg, ra1);               \
  st4(lB[buf] + sr0 * LPITCH + scg, rb0);               \
  st4(lB[buf] + sr1 * LPITCH + scg, rb1);

#define GEMM_COMPUTE(buf)                                                     \
  {                                                                           \
    v16bf af0 = load_frag(lA[buf] + (wm * 32 + lr) * LPITCH, lhalf);          \
    v16bf af1 = load_frag(lA[buf] + (wm * 32 + 16 + lr) * LPITCH, lhalf);     \
    v16bf bw0 = load_frag(lB[buf] + (wn * 64 + lr) * LPITCH, lhalf);          \
    v16bf bw1 = load_frag(lB[buf] + (wn * 64 + 16 + lr) * LPITCH, lhalf);     \
    v16bf bw2 = load_frag(lB[buf] + (wn * 64 + 32 + lr) * LPITCH, lhalf);     \
    v16bf bw3 = load_frag(lB[buf] + (wn * 64 + 48 + lr) * LPITCH, lhalf);     \
    acc00 = wmma_bf16(af0, bw0, acc00);                                       \
    acc01 = wmma_bf16(af0, bw1, acc01);                                       \
    acc02 = wmma_bf16(af0, bw2, acc02);                                       \
    acc03 = wmma_bf16(af0, bw3, acc03);                                       \
    acc10 = wmma_bf16(af1, bw0, acc10);                                       \
    acc11 = wmma_bf16(af1, bw1, acc11);                                       \
    acc12 = wmma_bf16(af1, bw2, acc12);                                       \
    acc13 = wmma_bf16(af1, bw3, acc13);                                       \
  }

#define GEMM_EPI(ACC, i, j)                                                   \
  {                                                                           \
    const int col = n0 + wn * 64 + (j)*16 + lr;                               \
    const float bv = bias ? bias[col] : 0.f;                                  \
    for (int v = 0; v < 8; ++v) {                                             \
      const int row = row0 + wm * 32 + (i)*16 + lhalf * 8 + v;                \
      if (FULL || row < R) {                                                  \
        float val = ACC[v] + bv;                                              \
        if (Cf) Cf[(size_t)row * N + col] = val;                              \
        if (Cb) Cb[(size_t)row * N + col] = f2bf(val);                        \
      }                                                                       \
    }                                                                         \
  }

template <bool FULL>
__global__ __launch_bounds__(256) void gemm_bf16_kernel(
    const unsigned short* __restrict__ A, const unsigned short* __restrict__ W,
    const float* __restrict__ bias, float* __restrict__ Cf,
    unsigned short* __restrict__ Cb, int R, int N, int K) {
  __shared__ alignas(16) unsigned short lA[2][GT_BM * LPITCH];
  __shared__ alignas(16) unsigned short lB[2][GT_BN * LPITCH];

  const int tid   = threadIdx.x;
  const int lane  = tid & 31;
  const int wave  = tid >> 5;
  const int wm    = wave >> 1;   // 0..3 -> 32-row band
  const int wn    = wave & 1;    // 0..1 -> 64-col band
  const int lhalf = lane >> 4;
  const int lr    = lane & 15;
  const int row0  = blockIdx.x * GT_BM;
  const int n0    = blockIdx.y * GT_BN;

  // staging indices: chunks c = tid, tid+256 of 512 uint4 per tile
  const int sr0 = tid >> 2, scg = (tid & 3) * 8;   // chunk tid
  const int sr1 = sr0 + 64;                        // chunk tid+256

  const bool okA0 = FULL || (row0 + sr0 < R);
  const bool okA1 = FULL || (row0 + sr1 < R);
  const uint4 uz = make_uint4(0u, 0u, 0u, 0u);

  const unsigned short* Ab0 = A + (size_t)(row0 + sr0) * K + scg;
  const unsigned short* Ab1 = A + (size_t)(row0 + sr1) * K + scg;
  const unsigned short* Wb0 = W + (size_t)(n0 + sr0) * K + scg;
  const unsigned short* Wb1 = W + (size_t)(n0 + sr1) * K + scg;

  v8f acc00{}, acc01{}, acc02{}, acc03{}, acc10{}, acc11{}, acc12{}, acc13{};
  uint4 ra0, ra1, rb0, rb1;

  const int nk = K >> 5;
  GEMM_LOAD(0);
  GEMM_STORE(0);

  for (int kk = 0; kk < nk - 1; ++kk) {
    __syncthreads();  // buf[kk&1] visible; all reads of buf[(kk+1)&1] retired
    GEMM_LOAD((kk + 1) << 5);   // unconditional steady-state prefetch
    const int buf = kk & 1;
    GEMM_COMPUTE(buf);
    GEMM_STORE(buf ^ 1);
  }
  __syncthreads();
  GEMM_COMPUTE((nk - 1) & 1);   // peeled last chunk

  GEMM_EPI(acc00, 0, 0); GEMM_EPI(acc01, 0, 1);
  GEMM_EPI(acc02, 0, 2); GEMM_EPI(acc03, 0, 3);
  GEMM_EPI(acc10, 1, 0); GEMM_EPI(acc11, 1, 1);
  GEMM_EPI(acc12, 1, 2); GEMM_EPI(acc13, 1, 3);
}

// ---------------------------------------------------------------------------
// c0 = broadcast(in_c); also resets the device-wide barrier.
// ---------------------------------------------------------------------------
__global__ void init_c_kernel(const float* __restrict__ in_c, float* c0f,
                              unsigned short* c0b, unsigned* bar, int Bsz, int M) {
  int idx = blockIdx.x * blockDim.x + threadIdx.x;
  if (idx == 0) { bar[0] = 0u; bar[1] = 0u; }
  if (idx < Bsz * M) {
    float v = in_c[idx % M];
    c0f[idx] = v;
    c0b[idx] = f2bf(v);
  }
}

// ---------------------------------------------------------------------------
// Persistent recurrence: grid (B/16, M/16) one-wave blocks, T steps with a
// device-wide generation barrier; c ping-pongs via explicit pointer swaps.
// Per step each block fuses the three gate WMMA chains + sigmoid/tanh cell
// update; named-scalar register staging pipelines loads under the WMMAs.
// ---------------------------------------------------------------------------
#define SCAN_LOAD(k0)                                    \
  pA0 = ldg4(aB0 + (k0)); pA1 = ldg4(aB1 + (k0));        \
  pF0 = ldg4(fB0 + (k0)); pF1 = ldg4(fB1 + (k0));        \
  pI0 = ldg4(iB0 + (k0)); pI1 = ldg4(iB1 + (k0));        \
  pC0 = ldg4(cB0 + (k0)); pC1 = ldg4(cB1 + (k0));

#define SCAN_STORE()                                     \
  st4(sA0, pA0); st4(sA1, pA1);                          \
  st4(sF0, pF0); st4(sF1, pF1);                          \
  st4(sI0, pI0); st4(sI1, pI1);                          \
  st4(sC0, pC0); st4(sC1, pC1);

#define SCAN_COMPUTE()                                   \
  {                                                      \
    v16bf av = load_frag(lA + lr * LPITCH, lhalf);       \
    v16bf fw = load_frag(lF + lr * LPITCH, lhalf);       \
    v16bf iw = load_frag(lI + lr * LPITCH, lhalf);       \
    v16bf cw = load_frag(lC + lr * LPITCH, lhalf);       \
    aF = wmma_bf16(av, fw, aF);                          \
    aI = wmma_bf16(av, iw, aI);                          \
    aC = wmma_bf16(av, cw, aC);                          \
  }

__global__ __launch_bounds__(32) void lstm_scan_kernel(
    const unsigned short* __restrict__ gxf, const unsigned short* __restrict__ gxi,
    const unsigned short* __restrict__ gxc, const unsigned short* __restrict__ Uf,
    const unsigned short* __restrict__ Ui, const unsigned short* __restrict__ Uc,
    const float* __restrict__ bf_, const float* __restrict__ bi_,
    const float* __restrict__ bc_, float* c0f, float* c1f, unsigned short* c0b,
    unsigned short* c1b, unsigned* bar, int T, int Bsz, int M) {
  __shared__ alignas(16) unsigned short lA[16 * LPITCH];
  __shared__ alignas(16) unsigned short lF[16 * LPITCH];
  __shared__ alignas(16) unsigned short lI[16 * LPITCH];
  __shared__ alignas(16) unsigned short lC[16 * LPITCH];

  const int lane  = threadIdx.x;
  const int lhalf = lane >> 4;
  const int lr    = lane & 15;
  const int row0  = blockIdx.x * 16;   // batch tile
  const int n0    = blockIdx.y * 16;   // M tile
  const unsigned nblocks = gridDim.x * gridDim.y;

  const float biasF = bf_[n0 + lr];
  const float biasI = bi_[n0 + lr];
  const float biasC = bc_[n0 + lr];

  // staging indices for the two uint4 chunks this lane owns per 16x32 tile
  const int r0 = lane >> 2, cg0 = (lane & 3) * 8;        // chunk = lane
  const int r1 = r0 + 8;                                 // chunk = lane + 32

  // constant base pointers (U weights + LDS slots)
  const unsigned short* fB0 = Uf + (size_t)(n0 + r0) * M + cg0;
  const unsigned short* fB1 = Uf + (size_t)(n0 + r1) * M + cg0;
  const unsigned short* iB0 = Ui + (size_t)(n0 + r0) * M + cg0;
  const unsigned short* iB1 = Ui + (size_t)(n0 + r1) * M + cg0;
  const unsigned short* cB0 = Uc + (size_t)(n0 + r0) * M + cg0;
  const unsigned short* cB1 = Uc + (size_t)(n0 + r1) * M + cg0;
  unsigned short* sA0 = lA + r0 * LPITCH + cg0;
  unsigned short* sA1 = lA + r1 * LPITCH + cg0;
  unsigned short* sF0 = lF + r0 * LPITCH + cg0;
  unsigned short* sF1 = lF + r1 * LPITCH + cg0;
  unsigned short* sI0 = lI + r0 * LPITCH + cg0;
  unsigned short* sI1 = lI + r1 * LPITCH + cg0;
  unsigned short* sC0 = lC + r0 * LPITCH + cg0;
  unsigned short* sC1 = lC + r1 * LPITCH + cg0;

  // explicit ping-pong pointers (no arrays -> address space stays global)
  const float*          curF = c0f;
  float*                nxtF = c1f;
  const unsigned short* curB = c0b;
  unsigned short*       nxtB = c1b;

  for (int t = 0; t < T; ++t) {
    // Init accumulators straight into WMMA C layout from precomputed x-projections.
    v8f aF, aI, aC;
    {
      const size_t g = ((size_t)t * Bsz + row0) * M + n0;
      for (int v = 0; v < 8; ++v) {
        size_t idx = g + (size_t)(lhalf * 8 + v) * M + lr;
        aF[v] = bf2f(gxf[idx]) + biasF;
        aI[v] = bf2f(gxi[idx]) + biasI;
        aC[v] = bf2f(gxc[idx]) + biasC;
      }
      if (t + 1 < T) {  // prefetch next step's pre-activation rows
        const size_t g2 = ((size_t)(t + 1) * Bsz + row0) * M + n0 + (size_t)lr * M;
        __builtin_prefetch(gxf + g2, 0, 3);
        __builtin_prefetch(gxi + g2, 0, 3);
        __builtin_prefetch(gxc + g2, 0, 3);
      }
    }

    const unsigned short* aB0 = curB + (size_t)(row0 + r0) * M + cg0;
    const unsigned short* aB1 = curB + (size_t)(row0 + r1) * M + cg0;

    uint4 pA0, pA1, pF0, pF1, pI0, pI1, pC0, pC1;
    SCAN_LOAD(0);
    for (int k0 = 0; k0 < M - 32; k0 += 32) {
      SCAN_STORE();
      __syncthreads();
      SCAN_LOAD(k0 + 32);      // unconditional prefetch overlaps the WMMAs
      SCAN_COMPUTE();
      __syncthreads();
    }
    SCAN_STORE();
    __syncthreads();
    SCAN_COMPUTE();            // peeled last chunk

    for (int v = 0; v < 8; ++v) {  // gates + cell update
      size_t ci = (size_t)(row0 + lhalf * 8 + v) * M + n0 + lr;
      float fg = 1.f / (1.f + __expf(-aF[v]));
      float ig = 1.f / (1.f + __expf(-aI[v]));
      float gg = tanhf(aC[v]);
      float cn = fg * curF[ci] + ig * gg;
      nxtF[ci] = cn;
      nxtB[ci] = f2bf(cn);
    }

    // Device-wide generation barrier (one per step; reads of cur precede it).
    __threadfence();
    __syncthreads();
    if (lane == 0) {
      volatile unsigned* vb = (volatile unsigned*)bar;
      unsigned gen = vb[1];
      __threadfence();
      unsigned prev = atomicAdd(bar, 1u);
      if (prev + 1u == nblocks) {
        vb[0] = 0u;
        __threadfence();
        vb[1] = gen + 1u;
      } else {
        while (vb[1] == gen) __builtin_amdgcn_s_sleep(1);
        __threadfence();
      }
    }
    __syncthreads();

    // swap ping-pong buffers
    {
      const float* tf = curF; curF = nxtF; nxtF = (float*)tf;
      const unsigned short* tb = curB; curB = nxtB; nxtB = (unsigned short*)tb;
    }
  }
}

// ---------------------------------------------------------------------------
// h[t,b,o] = tanh(GXo[t,b,o] + hc[b,o])   (GXo already holds x@wo.T + bo)
// ---------------------------------------------------------------------------
__global__ void output_kernel(const float* __restrict__ gxo,
                              const float* __restrict__ hc, float* __restrict__ out,
                              long long total, int Bsz, int O) {
  long long i = (long long)blockIdx.x * blockDim.x + threadIdx.x;
  long long stride = (long long)gridDim.x * blockDim.x;
  for (; i < total; i += stride) {
    int o = (int)(i % O);
    int b = (int)((i / O) % Bsz);
    out[i] = tanhf(gxo[i] + hc[(size_t)b * O + o]);
  }
}

// ---------------------------------------------------------------------------
extern "C" void kernel_launch(void* const* d_in, const int* in_sizes, int n_in,
                              void* d_out, int out_size, void* d_ws, size_t ws_size,
                              hipStream_t stream) {
  constexpr int T = 256, Bsz = 64, I = 1024, M = 1024, O = 1024;
  constexpr long long TB = (long long)T * Bsz;

  const float* x    = (const float*)d_in[0];
  const float* wf   = (const float*)d_in[1];
  const float* uf   = (const float*)d_in[2];
  const float* bf_b = (const float*)d_in[3];
  const float* wi   = (const float*)d_in[4];
  const float* ui   = (const float*)d_in[5];
  const float* bi_b = (const float*)d_in[6];
  const float* wc   = (const float*)d_in[7];
  const float* uc   = (const float*)d_in[8];
  const float* bc_b = (const float*)d_in[9];
  const float* wo   = (const float*)d_in[10];
  const float* uo   = (const float*)d_in[11];
  const float* bo   = (const float*)d_in[12];
  const float* in_c = (const float*)d_in[13];
  float* out = (float*)d_out;

  char* base = (char*)d_ws;
  size_t off = 0;
  auto alloc = [&](size_t bytes) -> void* {
    void* p = base + off;
    off += (bytes + 255) & ~(size_t)255;
    return p;
  };

  unsigned short* xb  = (unsigned short*)alloc((size_t)TB * I * 2);
  unsigned short* wfb = (unsigned short*)alloc((size_t)M * I * 2);
  unsigned short* ufb = (unsigned short*)alloc((size_t)M * M * 2);
  unsigned short* wib = (unsigned short*)alloc((size_t)M * I * 2);
  unsigned short* uib = (unsigned short*)alloc((size_t)M * M * 2);
  unsigned short* wcb = (unsigned short*)alloc((size_t)M * I * 2);
  unsigned short* ucb = (unsigned short*)alloc((size_t)M * M * 2);
  unsigned short* wob = (unsigned short*)alloc((size_t)O * I * 2);
  unsigned short* uob = (unsigned short*)alloc((size_t)O * M * 2);
  unsigned short* gxf = (unsigned short*)alloc((size_t)TB * M * 2);
  unsigned short* gxi = (unsigned short*)alloc((size_t)TB * M * 2);
  unsigned short* gxc = (unsigned short*)alloc((size_t)TB * M * 2);
  float*          gxo = (float*)alloc((size_t)TB * O * 4);
  float*          c0f = (float*)alloc((size_t)Bsz * M * 4);
  float*          c1f = (float*)alloc((size_t)Bsz * M * 4);
  unsigned short* c0b = (unsigned short*)alloc((size_t)Bsz * M * 2);
  unsigned short* c1b = (unsigned short*)alloc((size_t)Bsz * M * 2);
  float*          hc  = (float*)alloc((size_t)Bsz * O * 4);
  unsigned*       bar = (unsigned*)alloc(256);

  auto cvt = [&](const float* src, unsigned short* dst, long long n) {
    long long b = (n + 255) / 256;
    int blocks = (int)(b > 16384 ? 16384 : b);
    cvt_bf16_kernel<<<blocks, 256, 0, stream>>>(src, dst, n);
  };
  cvt(x, xb, TB * I);
  cvt(wf, wfb, (long long)M * I);
  cvt(uf, ufb, (long long)M * M);
  cvt(wi, wib, (long long)M * I);
  cvt(ui, uib, (long long)M * M);
  cvt(wc, wcb, (long long)M * I);
  cvt(uc, ucb, (long long)M * M);
  cvt(wo, wob, (long long)O * I);
  cvt(uo, uob, (long long)O * M);

  // Time-parallel input projections (gates in bf16; output branch f32 with +bo).
  dim3 gg((unsigned)(TB / GT_BM), (unsigned)(M / GT_BN));
  gemm_bf16_kernel<true><<<gg, 256, 0, stream>>>(xb, wfb, nullptr, nullptr, gxf,
                                                 (int)TB, M, I);
  gemm_bf16_kernel<true><<<gg, 256, 0, stream>>>(xb, wib, nullptr, nullptr, gxi,
                                                 (int)TB, M, I);
  gemm_bf16_kernel<true><<<gg, 256, 0, stream>>>(xb, wcb, nullptr, nullptr, gxc,
                                                 (int)TB, M, I);
  gemm_bf16_kernel<true><<<gg, 256, 0, stream>>>(xb, wob, bo, gxo, nullptr,
                                                 (int)TB, O, I);

  init_c_kernel<<<(Bsz * M + 255) / 256, 256, 0, stream>>>(in_c, c0f, c0b, bar,
                                                           Bsz, M);

  lstm_scan_kernel<<<dim3(Bsz / 16, M / 16), 32, 0, stream>>>(
      gxf, gxi, gxc, ufb, uib, ucb, bf_b, bi_b, bc_b, c0f, c1f, c0b, c1b, bar, T,
      Bsz, M);

  unsigned short* cfin = (T % 2 == 0) ? c0b : c1b;  // final cell state buffer
  dim3 gh(1, (unsigned)(O / GT_BN));
  gemm_bf16_kernel<false><<<gh, 256, 0, stream>>>(cfin, uob, nullptr, hc, nullptr,
                                                  Bsz, O, M);

  long long total = TB * O;
  int oblocks = (int)((total + 255) / 256);
  output_kernel<<<oblocks, 256, 0, stream>>>(gxo, hc, out, total, Bsz, O);
}
